// Trajectory2vecDecoder_8495445311952
// MI455X (gfx1250) — matrix operand order repaired and checked by reading, not compiled, not run on hardware
//
#include <hip/hip_runtime.h>
#include <hip/hip_bf16.h>
#include <math.h>

// Problem constants (from reference): B=64, L=64, H=128, S=64, IN=16
#define BB      64
#define LL      64
#define HH      128
#define SS      64
#define IIN     16
#define K2H     256    // 2H (GEMM K)
#define N4H     512    // 4H (GEMM N)
#define WGB     16     // batches per workgroup (GEMM M)
#define NTHREADS_A 512 // 16 wave32s

typedef __attribute__((ext_vector_type(16))) _Float16 v16h;
typedef __attribute__((ext_vector_type(8)))  _Float16 v8h;
typedef __attribute__((ext_vector_type(8)))  float    v8f;

__device__ __forceinline__ float sigmoidf_(float x) {
    return 1.0f / (1.0f + __expf(-x));
}

// ---------------------------------------------------------------------------
// Kernel A: 64-step LSTM recurrence over [16 batches/WG, H=128].
// Per step: g = [c,h](f16) @ W_gates(f16, resident in VGPRs) via WMMA,
// then f32 elementwise gate update in LDS. Writes c_all[s][b][h] (f32) to ws.
// ---------------------------------------------------------------------------
__global__ __launch_bounds__(NTHREADS_A)
void lstm_recurrence_kernel(const float* __restrict__ embedding, // [64][64][128]
                            const float* __restrict__ W_gates,   // [256][512]
                            const float* __restrict__ b_gates,   // [512]
                            float* __restrict__ c_all)           // [S][B][H]
{
    __shared__ _Float16 inp[WGB][K2H];   // activations [c,h] as f16 (8 KB)
    __shared__ float    gbuf[WGB][N4H];  // GEMM result (32 KB)
    __shared__ float    cst[WGB][HH];    // cell state f32 (8 KB)
    __shared__ float    bg[N4H];         // bias (2 KB)

    const int tid  = threadIdx.x;
    const int lane = tid & 31;
    const int wave = tid >> 5;                  // 0..15 -> N-tiles {2w, 2w+1}
    const int bg0  = blockIdx.x * WGB;          // batch base for this WG

    // --- init activations: inp = [embedding[b,0,:], zeros], c = 0 ---
    for (int i = tid; i < WGB * HH; i += NTHREADS_A) {
        int b = i >> 7, h = i & 127;
        float e = embedding[(size_t)(bg0 + b) * (LL * HH) + h];
        cst[b][h]       = 0.0f;
        inp[b][h]       = (_Float16)e;
        inp[b][HH + h]  = (_Float16)0.0f;
    }
    for (int i = tid; i < N4H; i += NTHREADS_A) bg[i] = b_gates[i];

    // --- one-time load of B (weight) fragments into VGPRs, f32 -> f16 ---
    // B k order used here (same as A, not the ISA table order): lanes 0-15: n=lane, K=0..15; lanes 16-31: n=lane-16, K=16..31
    const int ncol  = lane & 15;
    const int khalf = (lane >> 4) * 16;
    v16h bfrag[2][8];
    #pragma unroll
    for (int t = 0; t < 2; ++t) {
        const int n = (wave * 2 + t) * 16 + ncol;
        #pragma unroll
        for (int kc = 0; kc < 8; ++kc) {
            const int kb = kc * 32 + khalf;
            v16h bf;
            #pragma unroll
            for (int j = 0; j < 16; ++j)
                bf[j] = (_Float16)W_gates[(size_t)(kb + j) * N4H + n];
            bfrag[t][kc] = bf;
        }
    }
    __syncthreads();

    // A k order used here (same as B, not the ISA table order): lanes 0-15: m=lane, K=0..15;
    //                     lanes 16-31: m=lane-16, K=16..31
    const int am    = lane & 15;
    const int akoff = (lane >> 4) * 16;
    const int mo    = (lane >> 4) * 8;   // C/D row offset

    for (int s = 0; s < SS; ++s) {
        v8f acc0 = {}; v8f acc1 = {};
        #pragma unroll
        for (int kc = 0; kc < 8; ++kc) {
            const _Float16* ap = &inp[am][kc * 32 + akoff];
            v8h lo = *(const v8h*)(ap);        // ds_load_b128
            v8h hi = *(const v8h*)(ap + 8);    // ds_load_b128
            v16h a;
            #pragma unroll
            for (int j = 0; j < 8; ++j) { a[j] = lo[j]; a[8 + j] = hi[j]; }
            acc0 = __builtin_amdgcn_wmma_f32_16x16x32_f16(
                false, a, false, bfrag[0][kc], (short)0, acc0, false, false);
            acc1 = __builtin_amdgcn_wmma_f32_16x16x32_f16(
                false, a, false, bfrag[1][kc], (short)0, acc1, false, false);
        }
        // C/D 16x16 f32 layout: VGPR r -> row (r + 8*(lane>=16)), col lane%16
        #pragma unroll
        for (int r = 0; r < 8; ++r) {
            gbuf[mo + r][(wave * 2 + 0) * 16 + ncol] = acc0[r];
            gbuf[mo + r][(wave * 2 + 1) * 16 + ncol] = acc1[r];
        }
        __syncthreads();

        // elementwise gate update: 2048 (b,h) pairs / 512 threads = 4 each
        #pragma unroll
        for (int i = 0; i < 4; ++i) {
            const int p = tid + i * NTHREADS_A;
            const int b = p >> 7, h = p & 127;
            const float gf = gbuf[b][h]        + bg[h];
            const float gi = gbuf[b][HH + h]   + bg[HH + h];
            const float go = gbuf[b][2*HH + h] + bg[2*HH + h];
            const float gc = gbuf[b][3*HH + h] + bg[3*HH + h];
            const float cn = sigmoidf_(gf) * cst[b][h] + sigmoidf_(gi) * tanhf(gc);
            const float hn = sigmoidf_(go) * tanhf(cn);
            cst[b][h]      = cn;
            inp[b][h]      = (_Float16)cn;
            inp[b][HH + h] = (_Float16)hn;
            c_all[((size_t)s * BB + (bg0 + b)) * HH + h] = cn;
        }
        __syncthreads();
    }
}

// ---------------------------------------------------------------------------
// Kernel B: out[b, s*L + l, :] = c_all[s,b,:] @ W_lin + b_lin, replicated over l.
// Each block handles 16 (b,s) pairs; each pair's output is a contiguous
// 1024-float region = its 16-float projection repeated 64 times.
// ---------------------------------------------------------------------------
__global__ __launch_bounds__(256)
void projection_kernel(const float* __restrict__ c_all,  // [S][B][H]
                       const float* __restrict__ W_lin,  // [128][16]
                       const float* __restrict__ b_lin,  // [16]
                       float* __restrict__ out)          // [B][S*L][IN]
{
    __shared__ float wl[HH * IIN];   // 8 KB
    __shared__ float y[16][IIN];

    const int tid = threadIdx.x;
    for (int i = tid; i < HH * IIN; i += 256) wl[i] = W_lin[i];
    __syncthreads();

    const int pair0 = blockIdx.x * 16;       // global pair index p = b*S + s
    const int p   = pair0 + (tid >> 4);
    const int b   = p >> 6;
    const int s   = p & 63;
    const int col = tid & 15;

    const float* cv = &c_all[((size_t)s * BB + b) * HH];
    float acc = b_lin[col];
    #pragma unroll 8
    for (int k = 0; k < HH; ++k) acc += cv[k] * wl[k * IIN + col];
    y[tid >> 4][col] = acc;
    __syncthreads();

    #pragma unroll
    for (int r = 0; r < 16; ++r) {
        const int pr = pair0 + r;
        const int rb = pr >> 6, rs = pr & 63;
        float* obase = out + (size_t)rb * (SS * LL * IIN) + (size_t)rs * (LL * IIN);
        for (int i = tid; i < LL * IIN; i += 256)
            obase[i] = y[r][i & 15];
    }
}

// ---------------------------------------------------------------------------
extern "C" void kernel_launch(void* const* d_in, const int* in_sizes, int n_in,
                              void* d_out, int out_size, void* d_ws, size_t ws_size,
                              hipStream_t stream) {
    const float* trip      = (const float*)d_in[0];
    // d_in[1] = valid_len (unused by the reference computation)
    const float* embedding = (const float*)d_in[2];
    const float* W_gates   = (const float*)d_in[3];
    const float* b_gates   = (const float*)d_in[4];
    const float* W_lin     = (const float*)d_in[5];
    const float* b_lin     = (const float*)d_in[6];

    float* out   = (float*)d_out;
    float* c_all = (float*)d_ws;                 // [64][64][128] f32 = 2 MB

    // output 0: x = trip pass-through
    hipMemcpyAsync(out, trip, (size_t)in_sizes[0] * sizeof(float),
                   hipMemcpyDeviceToDevice, stream);

    // output 1: decoder states -> projection
    lstm_recurrence_kernel<<<BB / WGB, NTHREADS_A, 0, stream>>>(
        embedding, W_gates, b_gates, c_all);

    float* out1 = out + (size_t)in_sizes[0];     // 262144 floats in
    projection_kernel<<<(BB * SS) / 16, 256, 0, stream>>>(c_all, W_lin, b_lin, out1);
}
